// GraphConv_24352464568326
// MI455X (gfx1250) — matrix operand order repaired and checked by reading, not compiled, not run on hardware
//
#include <hip/hip_runtime.h>
#include <hip/hip_bf16.h>

#define N_ROWS  8192
#define IN_DIM  256
#define OUT_DIM 256

typedef __bf16 v16bf __attribute__((ext_vector_type(16)));
typedef float  v8f   __attribute__((ext_vector_type(8)));
typedef int    v4i   __attribute__((ext_vector_type(4)));

typedef __attribute__((address_space(1))) v4i gv4i;   // global
typedef __attribute__((address_space(3))) v4i lv4i;   // LDS

union FragAB { unsigned u[8]; v16bf v; };

// ---------------------------------------------------------------------------
// fp32 pair -> packed bf16 dword. Prefer HW v_cvt_pk_bf16_f32; else
// round-to-nearest (add 0x8000) + single v_perm_b32 to merge high halves.
// ---------------------------------------------------------------------------
#if __has_builtin(__builtin_amdgcn_cvt_pk_bf16_f32)
__device__ __forceinline__ unsigned pack2_bf16(float a, float b) {
    auto r = __builtin_amdgcn_cvt_pk_bf16_f32(a, b);
    union { decltype(r) v; unsigned u; } c;
    c.v = r;
    return c.u;
}
#else
__device__ __forceinline__ unsigned pack2_bf16(float a, float b) {
    unsigned ua = __float_as_uint(a) + 0x8000u;
    unsigned ub = __float_as_uint(b) + 0x8000u;
    // D[15:0] = ua[31:16], D[31:16] = ub[31:16]  (bytes: {ub3,ub2,ua3,ua2})
    return __builtin_amdgcn_perm(ub, ua, 0x07060302u);
}
#endif

// ---------------------------------------------------------------------------
// 16-byte global -> LDS copy: async (ASYNCcnt) when available, else sync.
// cpol: temporal hint for the global read (0 = RT, 1 = NT stream).
// ---------------------------------------------------------------------------
#if __has_builtin(__builtin_amdgcn_global_load_async_to_lds_b128)
#define HAVE_ASYNC_LDS 1
#endif

template <int CPOL>
__device__ __forceinline__ void copy16_g2l(const void* g, void* l) {
#ifdef HAVE_ASYNC_LDS
    __builtin_amdgcn_global_load_async_to_lds_b128(
        (gv4i*)(unsigned long long)g,
        (lv4i*)(unsigned)(unsigned long long)l, 0, CPOL);
#else
    *reinterpret_cast<uint4*>(l) = *reinterpret_cast<const uint4*>(g);
#endif
}

__device__ __forceinline__ void async_wait0() {
#ifdef HAVE_ASYNC_LDS
#if __has_builtin(__builtin_amdgcn_s_wait_asynccnt)
    __builtin_amdgcn_s_wait_asynccnt(0);
#else
    asm volatile("s_wait_asynccnt 0" ::: "memory");
#endif
#endif
}

// ---------------------------------------------------------------------------
// Pack fp32 row-major [K x ldn] into WMMA B-fragment tiles (bf16).
// Tile = 32(K) x 16(N); one 256-thread block per tile.
// B order used here (32x16, contiguous K halves): lane L -> n = L&15, K-half = L>>4,
// VGPR v holds B[k0+2v][n], B[k0+2v+1][n].
// ---------------------------------------------------------------------------
__global__ __launch_bounds__(256) void pack_b_tiles(
    const float* __restrict__ src, int ldn, int numNt,
    unsigned* __restrict__ dst)
{
    const int tile = blockIdx.x;
    const int kt   = tile / numNt;
    const int nt   = tile % numNt;
    const int lane = threadIdx.x >> 3;
    const int v    = threadIdx.x & 7;
    const int n    = nt * 16 + (lane & 15);
    const int k    = kt * 32 + (lane >> 4) * 16 + v * 2;
    const float f0 = src[(size_t)k * ldn + n];
    const float f1 = src[(size_t)(k + 1) * ldn + n];
    dst[(size_t)tile * 256 + lane * 8 + v] = pack2_bf16(f0, f1);
}

// ---------------------------------------------------------------------------
// GEMM1: agg = A[8192,8192] * feat[8192,256]  (bf16 WMMA, f32 acc)
// 256 thr = 8 waves as 4(M) x 2(N); block tile 64x256; K-step 32.
// Double-buffered LDS pipeline; async global->LDS staging (A tagged NT).
// ---------------------------------------------------------------------------
__global__ __launch_bounds__(256) void gemm1_agg(
    const float* __restrict__ A,
    const unsigned* __restrict__ featB,   // packed tiles: 256 kt x 16 nt
    float* __restrict__ agg)
{
    __shared__ float    As[2][64 * 36];   // padded stride 36 (conflict-free)
    __shared__ unsigned Bs[2][16 * 256];  // K-chunk: 16 ntiles x 32 x 8

    const int tid  = threadIdx.x;
    const int wave = tid >> 5;
    const int lane = tid & 31;
    const int h    = lane >> 4;
    const int ml   = lane & 15;
    const int wm   = wave & 3;
    const int wn   = wave >> 2;
    const int m0   = blockIdx.x * 64;

    auto stage = [&](int kt, int buf) {
        // A chunk: 64 rows x 32 fp32 = 512 x b128, coalesced, NT stream
#pragma unroll
        for (int j = 0; j < 2; ++j) {
            const int fidx = tid + j * 256;
            const int row  = fidx >> 3;
            const int c4   = fidx & 7;
            copy16_g2l<1>(&A[(size_t)(m0 + row) * N_ROWS + (size_t)kt * 32 + c4 * 4],
                          &As[buf][row * 36 + c4 * 4]);
        }
        // packed-B chunk: contiguous 16 KB, L2-resident (RT)
        const unsigned* src = featB + (size_t)kt * (16 * 256);
#pragma unroll
        for (int j = 0; j < 4; ++j) {
            const int idx = (tid + j * 256) * 4;
            copy16_g2l<0>(src + idx, &Bs[buf][idx]);
        }
    };

    v8f zero = {};
    v8f acc[8];
#pragma unroll
    for (int t = 0; t < 8; ++t) acc[t] = zero;

    stage(0, 0);
    async_wait0();
    __syncthreads();

    for (int kt = 0; kt < N_ROWS / 32; ++kt) {
        const int cur = kt & 1;
        if (kt + 1 < N_ROWS / 32) stage(kt + 1, cur ^ 1);   // overlap with compute

        // A fragment: 16x32 bf16 (contiguous-K order, same as packed B), cvt in-register
        FragAB a;
#pragma unroll
        for (int v = 0; v < 8; ++v) {
            const int kl = h * 16 + v * 2;
            float2 f = *reinterpret_cast<const float2*>(
                &As[cur][(wm * 16 + ml) * 36 + kl]);
            a.u[v] = pack2_bf16(f.x, f.y);
        }
#pragma unroll
        for (int t = 0; t < 8; ++t) {
            FragAB b;
            const uint4* bp = reinterpret_cast<const uint4*>(
                &Bs[cur][((wn * 8 + t) * 32 + lane) * 8]);
            *reinterpret_cast<uint4*>(&b.u[0]) = bp[0];
            *reinterpret_cast<uint4*>(&b.u[4]) = bp[1];
            acc[t] = __builtin_amdgcn_wmma_f32_16x16x32_bf16(
                false, a.v, false, b.v, (short)0, acc[t], false, false);
        }

        async_wait0();     // next-chunk staging landed
        __syncthreads();   // all waves done reading cur
    }

    // epilogue: C layout m = r + 8*h, n = ml
#pragma unroll
    for (int t = 0; t < 8; ++t) {
        const int n = wn * 128 + t * 16 + ml;
#pragma unroll
        for (int r = 0; r < 8; ++r) {
            const int m = m0 + wm * 16 + 8 * h + r;
            agg[(size_t)m * OUT_DIM + n] = acc[t][r];
        }
    }
}

// ---------------------------------------------------------------------------
// GEMM2: out = relu([feat, agg] @ W + bias), K = 512, fused epilogue.
// ---------------------------------------------------------------------------
__global__ __launch_bounds__(256) void gemm2_out(
    const float* __restrict__ feat,
    const float* __restrict__ agg,
    const unsigned* __restrict__ wpk,     // packed W tiles: 16 kt x 16 nt
    const float* __restrict__ bias,
    float* __restrict__ out)
{
    const int tid  = threadIdx.x;
    const int wave = tid >> 5;
    const int lane = tid & 31;
    const int h    = lane >> 4;
    const int ml   = lane & 15;
    const int wm   = wave & 3;
    const int wn   = wave >> 2;
    const int m0   = blockIdx.x * 64;
    const int row  = m0 + wm * 16 + ml;

    v8f zero = {};
    v8f acc[8];
#pragma unroll
    for (int t = 0; t < 8; ++t) acc[t] = zero;

#pragma unroll 1
    for (int kt = 0; kt < 16; ++kt) {
        FragAB a;
#pragma unroll
        for (int v = 0; v < 8; ++v) {
            const int kl = kt * 32 + h * 16 + v * 2;
            const float* sp = (kl < IN_DIM)
                ? &feat[(size_t)row * IN_DIM + kl]
                : &agg [(size_t)row * IN_DIM + (kl - IN_DIM)];
            float2 f = *reinterpret_cast<const float2*>(sp);
            a.u[v] = pack2_bf16(f.x, f.y);
        }
#pragma unroll
        for (int t = 0; t < 8; ++t) {
            FragAB b;
            const uint4* bp = reinterpret_cast<const uint4*>(
                &wpk[((size_t)(kt * 16 + wn * 8 + t) * 32 + lane) * 8]);
            *reinterpret_cast<uint4*>(&b.u[0]) = bp[0];
            *reinterpret_cast<uint4*>(&b.u[4]) = bp[1];
            acc[t] = __builtin_amdgcn_wmma_f32_16x16x32_bf16(
                false, a.v, false, b.v, (short)0, acc[t], false, false);
        }
    }

#pragma unroll
    for (int t = 0; t < 8; ++t) {
        const int n  = wn * 128 + t * 16 + ml;
        const float bv = bias[n];
#pragma unroll
        for (int r = 0; r < 8; ++r) {
            const int m = m0 + wm * 16 + 8 * h + r;
            float o = acc[t][r] + bv;
            out[(size_t)m * OUT_DIM + n] = o > 0.0f ? o : 0.0f;
        }
    }
}

// ---------------------------------------------------------------------------
extern "C" void kernel_launch(void* const* d_in, const int* in_sizes, int n_in,
                              void* d_out, int out_size, void* d_ws, size_t ws_size,
                              hipStream_t stream) {
    const float* feat = (const float*)d_in[0];   // [8192,256]
    const float* A    = (const float*)d_in[1];   // [8192,8192]
    const float* W    = (const float*)d_in[2];   // [512,256]
    const float* bias = (const float*)d_in[3];   // [256]
    float* out = (float*)d_out;

    char* ws = (char*)d_ws;
    unsigned* featB = (unsigned*)ws;                          // 4 MB packed features
    unsigned* wpk   = (unsigned*)(ws + (4u << 20));           // 256 KB packed weight
    float*    agg   = (float*)(ws + (4u << 20) + (1u << 18)); // 8 MB agg buffer

    pack_b_tiles<<<256 * 16, 256, 0, stream>>>(feat, IN_DIM, 16, featB);
    pack_b_tiles<<<16 * 16, 256, 0, stream>>>(W, OUT_DIM, 16, wpk);

    gemm1_agg<<<N_ROWS / 64, 256, 0, stream>>>(A, featB, agg);
    gemm2_out<<<N_ROWS / 64, 256, 0, stream>>>(feat, agg, wpk, bias, out);
}
